// ReidRNSelf_63307817943799
// MI455X (gfx1250) — matrix operand order repaired and checked by reading, not compiled, not run on hardware
//
#include <hip/hip_runtime.h>
#include <hip/hip_bf16.h>

// Problem constants (from the reference)
#define MBATCH 128
#define CCH    2048
#define PP     32       // H*W
#define DG     256
#define DF     256
#define DOUT_  128

typedef __attribute__((ext_vector_type(16))) __bf16 v16bf;
typedef __attribute__((ext_vector_type(8)))  __bf16 v8bf;
typedef __attribute__((ext_vector_type(8)))  float  v8f;

// float -> bf16 (round to nearest even), stored as ushort
__device__ __forceinline__ unsigned short f2bfu(float f) {
  unsigned u = __builtin_bit_cast(unsigned, f);
  unsigned r = (u + 0x7FFFu + ((u >> 16) & 1u)) >> 16;
  return (unsigned short)r;
}

__device__ __forceinline__ v16bf cat8(v8bf lo, v8bf hi) {
  union { v16bf v; v8bf h[2]; } u;
  u.h[0] = lo; u.h[1] = hi;
  return u.v;
}

// ---- CDNA5 async copy: global -> LDS, 16B per lane, ASYNCcnt-tracked ----
// Per ISA flat-aperture mapping, low 32 bits of a generic LDS pointer are the
// wave-relative LDS byte offset, which is exactly what VDST wants.
__device__ __forceinline__ void async_g2lds_b128(void* lds_ptr, const void* gptr) {
  unsigned lds_addr = (unsigned)(unsigned long long)lds_ptr;
  unsigned long long ga = (unsigned long long)gptr;
  asm volatile("global_load_async_to_lds_b128 %0, %1, off"
               :: "v"(lds_addr), "v"(ga) : "memory");
}

__device__ __forceinline__ void wait_async0() {
#if __has_builtin(__builtin_amdgcn_s_wait_asynccnt)
  __builtin_amdgcn_s_wait_asynccnt(0);
#else
  asm volatile("s_wait_asynccnt 0x0" ::: "memory");
#endif
}

// stage a 256x256 bf16 weight (128KB) into LDS: 8192 x 16B chunks, 32 per thread
__device__ __forceinline__ void stage_weight_async(const unsigned short* __restrict__ W,
                                                   unsigned short* wbuf, int tid) {
  #pragma unroll
  for (int k = 0; k < 32; ++k) {
    int j = tid + k * 256;                 // j in [0, 8192)
    async_g2lds_b128((char*)wbuf + (size_t)j * 16, (const char*)W + (size_t)j * 16);
  }
}

// ---------------- prep kernels (fp32 -> bf16, transposes) ----------------

// xbf[b][p][c] = bf16( x[b][c][p] )
__global__ void k_prep_x(const float* __restrict__ x, unsigned short* __restrict__ xbf) {
  int i = blockIdx.x * 256 + threadIdx.x;
  int c = i % CCH;
  int bp = i / CCH;
  int p = bp % PP;
  int b = bp / PP;
  xbf[i] = f2bfu(x[((size_t)b * CCH + c) * PP + p]);
}

// W1T[d][c], d in [0,512): d<256 -> Wg1[:2048] col d ; d>=256 -> Wg1[2048:] col d-256
__global__ void k_prep_w1(const float* __restrict__ Wg1, unsigned short* __restrict__ W1T) {
  int i = blockIdx.x * 256 + threadIdx.x;   // N = 512*2048
  int c = i % CCH;
  int d = i / CCH;
  float v = (d < DG) ? Wg1[(size_t)c * DG + d]
                     : Wg1[(size_t)(CCH + c) * DG + (d - DG)];
  W1T[i] = f2bfu(v);
}

// WT[d][c] = bf16( W[c][d] ) for 256x256 weights
__global__ void k_transpose256(const float* __restrict__ W, unsigned short* __restrict__ WT) {
  int i = blockIdx.x * 256 + threadIdx.x;   // N = 65536
  int c = i % DG;
  int d = i / DG;
  WT[i] = f2bfu(W[(size_t)c * DG + d]);
}

__global__ void k_zero(float* __restrict__ p, int n) {
  int i = blockIdx.x * 256 + threadIdx.x;
  if (i < n) p[i] = 0.f;
}

// ---------------- kernel 1: A = xT@W1a, B = xT@W1b  (bf16 WMMA, fp32 acc) ----------------
// grid (4, MBATCH), block 256. wave w handles n-tile (blockIdx.x*8 + w), 16 cols, M = 32 rows.
__global__ void __launch_bounds__(256) k_gemm1(
    const unsigned short* __restrict__ xbf, const unsigned short* __restrict__ W1T,
    float* __restrict__ Aall, float* __restrict__ Ball) {
  int b = blockIdx.y;
  int w = threadIdx.x >> 5;
  int l = threadIdx.x & 31;
  int l15 = l & 15;
  int lhalf = l >> 4;
  int nt = blockIdx.x * 8 + w;                    // 0..31 over 512 cols
  const unsigned short* abase = xbf + (size_t)b * PP * CCH;
  const unsigned short* arow0 = abase + (size_t)l15 * CCH;
  const unsigned short* arow1 = abase + (size_t)(16 + l15) * CCH;
  const unsigned short* brow  = W1T + (size_t)(nt * 16 + l15) * CCH;

  v8f acc0 = {0,0,0,0,0,0,0,0};
  v8f acc1 = {0,0,0,0,0,0,0,0};
  for (int kc = 0; kc < CCH / 32; ++kc) {
    int ka = kc * 32 + (lhalf << 3);              // A: interleaved K layout
    v16bf a0 = cat8(*(const v8bf*)(arow0 + ka), *(const v8bf*)(arow0 + ka + 16));
    v16bf a1 = cat8(*(const v8bf*)(arow1 + ka), *(const v8bf*)(arow1 + ka + 16));
    int kb = kc * 32 + (lhalf << 3);              // B: interleaved K layout (same as A)
    v16bf bb = cat8(*(const v8bf*)(brow + kb), *(const v8bf*)(brow + kb + 16));
    acc0 = __builtin_amdgcn_wmma_f32_16x16x32_bf16(false, a0, false, bb, (short)0, acc0, false, false);
    acc1 = __builtin_amdgcn_wmma_f32_16x16x32_bf16(false, a1, false, bb, (short)0, acc1, false, false);
  }
  int d = nt * 16 + l15;
  float* dst; int dcol;
  if (d < DG) { dst = Aall; dcol = d; } else { dst = Ball; dcol = d - DG; }
  int mofs = lhalf * 8;
  #pragma unroll
  for (int e = 0; e < 8; ++e) {
    int p0 = e + mofs;                            // D layout: lanes>=16 hold M=r+8
    dst[((size_t)b * PP + p0) * DG + dcol]      = acc0[e];
    dst[((size_t)b * PP + 16 + p0) * DG + dcol] = acc1[e];
  }
}

// ---------------- kernel 2: fused pairwise MLP + reduction ----------------
// grid (8, MBATCH): blockIdx.x = 128-row tile of the 1024 (q,p) rows; blockIdx.y = batch.
// 8 waves, each owns 16 rows (wave-private in LDS -> no barriers on hbuf).
// Weights double-buffered in LDS via async global->LDS copies so layer L+1's
// 128KB stages while layer L's 128 WMMAs/wave run. LDS use = 320KB (WGP max).
__global__ void __launch_bounds__(256) k_pairmlp(
    const float* __restrict__ Aall, const float* __restrict__ Ball,
    const float* __restrict__ bg1,
    const unsigned short* __restrict__ W2T, const float* __restrict__ bg2,
    const unsigned short* __restrict__ W3T, const float* __restrict__ bg3,
    const unsigned short* __restrict__ W4T, const float* __restrict__ bg4,
    float* __restrict__ xg) {
  __shared__ __align__(16) unsigned short hbuf[128 * DG];   //  64 KB activations
  __shared__ __align__(16) unsigned short wbuf0[DG * DG];   // 128 KB weight buffer 0
  __shared__ __align__(16) unsigned short wbuf1[DG * DG];   // 128 KB weight buffer 1

  int b    = blockIdx.y;
  int tile = blockIdx.x;
  int tid  = threadIdx.x;
  int w = tid >> 5, l = tid & 31, l15 = l & 15, lhalf = l >> 4;

  // Kick off async stage of layer-2 weights immediately; overlaps with Phase A.
  stage_weight_async(W2T, wbuf0, tid);

  // Phase A: h1 rows for this wave: relu(A[b,p,:] + B[b,q,:] + bg1)
  for (int i = 0; i < 16; ++i) {
    int r = tile * 128 + w * 16 + i;
    int q = r >> 5, p = r & 31;
    const float* arow = Aall + ((size_t)b * PP + p) * DG;
    const float* brow = Ball + ((size_t)b * PP + q) * DG;
    unsigned short* hrow = hbuf + (w * 16 + i) * DG;
    for (int col = l; col < DG; col += 32) {
      float v = arow[col] + brow[col] + bg1[col];
      hrow[col] = f2bfu(v > 0.f ? v : 0.f);
    }
  }
  wait_async0();
  __syncthreads();                                // wbuf0 ready, everyone here

  float xacc[16];
  #pragma unroll
  for (int nt = 0; nt < 16; ++nt) xacc[nt] = 0.f;

  const unsigned short* WTs[3] = {W2T, W3T, W4T};
  const float* bgs[3] = {bg2, bg3, bg4};

  for (int L = 0; L < 3; ++L) {
    const unsigned short* wl = (L & 1) ? wbuf1 : wbuf0;
    unsigned short* wn       = (L & 1) ? wbuf0 : wbuf1;
    if (L < 2) stage_weight_async(WTs[L + 1], wn, tid);   // overlap with compute

    v8f acc[16];
    #pragma unroll
    for (int nt = 0; nt < 16; ++nt) acc[nt] = (v8f){0,0,0,0,0,0,0,0};

    const unsigned short* arow = hbuf + (w * 16 + l15) * DG;
    for (int kc = 0; kc < DG / 32; ++kc) {
      int ka = kc * 32 + (lhalf << 3);
      v16bf af = cat8(*(const v8bf*)(arow + ka), *(const v8bf*)(arow + ka + 16));
      int kb = kc * 32 + (lhalf << 3);
      #pragma unroll
      for (int nt = 0; nt < 16; ++nt) {
        const unsigned short* brow = wl + (nt * 16 + l15) * DG;
        v16bf bf_ = cat8(*(const v8bf*)(brow + kb), *(const v8bf*)(brow + kb + 16));
        acc[nt] = __builtin_amdgcn_wmma_f32_16x16x32_bf16(false, af, false, bf_, (short)0, acc[nt], false, false);
      }
    }

    if (L < 2) {
      // relu(acc + bias) -> back into this wave's hbuf rows as bf16
      #pragma unroll
      for (int nt = 0; nt < 16; ++nt) {
        float bv = bgs[L][nt * 16 + l15];
        #pragma unroll
        for (int e = 0; e < 8; ++e) {
          int m = e + lhalf * 8;
          float v = acc[nt][e] + bv;
          hbuf[(w * 16 + m) * DG + nt * 16 + l15] = f2bfu(v > 0.f ? v : 0.f);
        }
      }
    } else {
      // layer 4: fold row-sum directly out of the D fragments
      #pragma unroll
      for (int nt = 0; nt < 16; ++nt) {
        float bv = bgs[L][nt * 16 + l15];
        float s = 0.f;
        #pragma unroll
        for (int e = 0; e < 8; ++e) {
          float v = acc[nt][e] + bv;
          s += (v > 0.f ? v : 0.f);
        }
        xacc[nt] += s;
      }
    }

    wait_async0();      // this wave's async stage for L+1 landed
    __syncthreads();    // all waves done with wl and all stages visible
  }

  // block-level (q,p) reduction straight to global (L2-resident, 256 addrs)
  #pragma unroll
  for (int nt = 0; nt < 16; ++nt)
    atomicAdd(&xg[(size_t)b * DG + nt * 16 + l15], xacc[nt]);
}

// ---------------- kernel 3: tiny head MLP, fp32 ----------------
__global__ void __launch_bounds__(256) k_head(
    const float* __restrict__ xg,
    const float* __restrict__ Wf1,  const float* __restrict__ bf1,
    const float* __restrict__ Wfc2, const float* __restrict__ bfc2,
    const float* __restrict__ Wfc3, const float* __restrict__ bfc3,
    float* __restrict__ out) {
  __shared__ float s0[DG], s1[DF];
  int b = blockIdx.x, tid = threadIdx.x;
  s0[tid] = xg[(size_t)b * DG + tid];
  __syncthreads();
  float a = bf1[tid];
  for (int k = 0; k < DG; ++k) a += s0[k] * Wf1[(size_t)k * DF + tid];
  s1[tid] = a > 0.f ? a : 0.f;
  __syncthreads();
  float c = bfc2[tid];
  for (int k = 0; k < DF; ++k) c += s1[k] * Wfc2[(size_t)k * DF + tid];
  float rc = c > 0.f ? c : 0.f;
  __syncthreads();
  s0[tid] = rc;
  __syncthreads();
  if (tid < DOUT_) {
    float o = bfc3[tid];
    for (int k = 0; k < DF; ++k) o += s0[k] * Wfc3[(size_t)k * DOUT_ + tid];
    out[(size_t)b * DOUT_ + tid] = o;
  }
}

// ---------------- launch ----------------
extern "C" void kernel_launch(void* const* d_in, const int* in_sizes, int n_in,
                              void* d_out, int out_size, void* d_ws, size_t ws_size,
                              hipStream_t stream) {
  (void)in_sizes; (void)n_in; (void)out_size; (void)ws_size;
  const float* x    = (const float*)d_in[0];
  const float* Wg1  = (const float*)d_in[1];
  const float* bg1  = (const float*)d_in[2];
  const float* Wg2  = (const float*)d_in[3];
  const float* bg2  = (const float*)d_in[4];
  const float* Wg3  = (const float*)d_in[5];
  const float* bg3  = (const float*)d_in[6];
  const float* Wg4  = (const float*)d_in[7];
  const float* bg4  = (const float*)d_in[8];
  const float* Wf1  = (const float*)d_in[9];
  const float* bf1  = (const float*)d_in[10];
  const float* Wfc2 = (const float*)d_in[11];
  const float* bfc2 = (const float*)d_in[12];
  const float* Wfc3 = (const float*)d_in[13];
  const float* bfc3 = (const float*)d_in[14];

  char* ws = (char*)d_ws;
  size_t off = 0;
  auto carve = [&](size_t bytes) {
    char* p = ws + off;
    off += (bytes + 255) & ~(size_t)255;
    return p;
  };
  unsigned short* xbf = (unsigned short*)carve((size_t)MBATCH * PP * CCH * 2);  // 16 MB
  unsigned short* W1T = (unsigned short*)carve((size_t)(2 * DG) * CCH * 2);     //  2 MB
  unsigned short* W2T = (unsigned short*)carve((size_t)DG * DG * 2);
  unsigned short* W3T = (unsigned short*)carve((size_t)DG * DG * 2);
  unsigned short* W4T = (unsigned short*)carve((size_t)DG * DG * 2);
  float* Aall = (float*)carve((size_t)MBATCH * PP * DG * 4);                    //  4 MB
  float* Ball = (float*)carve((size_t)MBATCH * PP * DG * 4);                    //  4 MB
  float* xg   = (float*)carve((size_t)MBATCH * DG * 4);

  k_prep_x<<<(MBATCH * PP * CCH) / 256, 256, 0, stream>>>(x, xbf);
  k_prep_w1<<<((2 * DG) * CCH) / 256, 256, 0, stream>>>(Wg1, W1T);
  k_transpose256<<<(DG * DG) / 256, 256, 0, stream>>>(Wg2, W2T);
  k_transpose256<<<(DG * DG) / 256, 256, 0, stream>>>(Wg3, W3T);
  k_transpose256<<<(DG * DG) / 256, 256, 0, stream>>>(Wg4, W4T);
  k_zero<<<(MBATCH * DG) / 256, 256, 0, stream>>>(xg, MBATCH * DG);

  k_gemm1<<<dim3(4, MBATCH), 256, 0, stream>>>(xbf, W1T, Aall, Ball);
  k_pairmlp<<<dim3(8, MBATCH), 256, 0, stream>>>(Aall, Ball, bg1,
                                                 W2T, bg2, W3T, bg3, W4T, bg4, xg);
  k_head<<<MBATCH, 256, 0, stream>>>(xg, Wf1, bf1, Wfc2, bfc2, Wfc3, bfc3, (float*)d_out);
}
